// SwinSTTransformerModule_16535624090234
// MI455X (gfx1250) — matrix operand order repaired and checked by reading, not compiled, not run on hardware
//
#include <hip/hip_runtime.h>
#include <hip/hip_bf16.h>
#include <cstdint>

// Problem constants (match reference)
#define BATCH 8
#define CCH   64
#define TT    16
#define VV    207
#define NHEAD 8
#define DHD   8
#define WSZ_  4
#define NWD   4
#define NWIN  828                 // WSZ * V
#define MROWS (BATCH*TT*VV)       // 26496, divisible by 128
#define TOTE  (MROWS*CCH)         // 1,695,744

typedef __attribute__((ext_vector_type(16))) _Float16 v16h;
typedef __attribute__((ext_vector_type(8)))  _Float16 v8h;
typedef __attribute__((ext_vector_type(2)))  _Float16 v2h;
typedef __attribute__((ext_vector_type(8)))  float    v8f;

__device__ __forceinline__ v8f wmma_f32_f16(v16h a, v16h b, v8f c) {
  return __builtin_amdgcn_wmma_f32_16x16x32_f16(false, a, false, b, (short)0, c,
                                                false, false);
}
// A-matrix (16x32 f16) K index for VGPR pair vi, lane-half hf (contiguous: k = 16*hf + 2*vi)
__device__ __forceinline__ int akk(int vi, int hf) {
  return 16*hf + 2*vi;
}

// ---------------- layout transposes -----------------------------------------
__global__ void k_to_tvc(const float* __restrict__ x, float* __restrict__ cur) {
  int i = blockIdx.x * blockDim.x + threadIdx.x;
  if (i >= TOTE) return;
  int c  = i & (CCH-1);
  int tv = i >> 6;
  int v  = tv % VV;
  int bt = tv / VV;
  int t  = bt & (TT-1);
  int b  = bt >> 4;
  cur[i] = x[((b*CCH + c)*TT + t)*VV + v];
}

__global__ void k_to_ctv(const float* __restrict__ cur, float* __restrict__ out) {
  int i = blockIdx.x * blockDim.x + threadIdx.x;
  if (i >= TOTE) return;
  int v = i % VV;
  int t = (i / VV) & (TT-1);
  int c = (i / (VV*TT)) & (CCH-1);
  int b = i / (VV*TT*CCH);
  out[i] = cur[((b*TT + t)*VV + v)*CCH + c];
}

// ---------------- WMMA GEMM: out = A[M,KDIM] @ W[KDIM,N] + bias (+resid) -----
// 8 waves/block, one 16x16 tile/wave; shared KDIMx16 B-panel staged into LDS
// once per block via async global->LDS DMA (CDNA5 path, ASYNCcnt-tracked).
template<int KDIM, bool RESID, bool HALFOUT>
__global__ void k_gemm(const float* __restrict__ A, const float* __restrict__ W,
                       const float* __restrict__ bias, const float* __restrict__ resid,
                       float* __restrict__ outF, _Float16* __restrict__ outH,
                       int N) {
  __shared__ float bstage[KDIM * 16];   // 4 KB for KDIM=64
  const int tid  = threadIdx.x;         // 256 threads
  const int lane = tid & 31;
  const int wave = tid >> 5;
  const int hf   = lane >> 4;
  const int l15  = lane & 15;
  const int m0   = (blockIdx.x * 8 + wave) * 16;
  const int n0   = blockIdx.y * 16;
  const int mrow = m0 + l15;
  const int ncol = n0 + l15;

  // Async-stage the B panel: one 16-byte chunk per thread (KDIM*16*4 = 4096 B).
  {
    const int krow = tid >> 2;              // 0..63
    const int cch  = (tid & 3) * 4;         // 0,4,8,12
    const float* gsrc = W + krow * N + n0 + cch;              // 16B-aligned
    const unsigned ldsoff = (unsigned)(uintptr_t)(&bstage[krow * 16 + cch]);
    const unsigned long long gaddr = (unsigned long long)(uintptr_t)gsrc;
    asm volatile("global_load_async_to_lds_b128 %0, %1, off"
                 :: "v"(ldsoff), "v"(gaddr) : "memory");
  }
  __builtin_prefetch(&A[mrow * KDIM], 0, 1);      // warm A row while DMA runs
  asm volatile("s_wait_asynccnt 0x0" ::: "memory");
  __syncthreads();

  v8f acc = {};
#pragma unroll
  for (int k0 = 0; k0 < KDIM; k0 += 32) {
    v16h a, b;
#pragma unroll
    for (int vi = 0; vi < 8; ++vi) {
      const int ka = k0 + akk(vi, hf);
      const float2 av = *(const float2*)&A[mrow * KDIM + ka];   // 8B-aligned
      a[2*vi]   = (_Float16)av.x;
      a[2*vi+1] = (_Float16)av.y;
      const int kb = k0 + 16*hf + 2*vi;
      b[2*vi]   = (_Float16)bstage[kb * 16 + l15];
      b[2*vi+1] = (_Float16)bstage[(kb + 1) * 16 + l15];
    }
    acc = wmma_f32_f16(a, b, acc);
  }
  const float bv = bias[ncol];
#pragma unroll
  for (int r = 0; r < 8; ++r) {
    const int mr = m0 + r + 8*hf;
    float val = acc[r] + bv;
    if (RESID)   val += resid[mr * N + ncol];
    if (HALFOUT) outH[mr * N + ncol] = (_Float16)val;
    else         outF[mr * N + ncol] = val;
  }
}

// ---------------- windowed attention (flash-style, WMMA) ---------------------
// qkv: [MROWS, 192] f16 (q|k|v, each H*DH=64). out: [MROWS, 64] f32.
// block = 128 threads (4 waves); wave -> one 16-row Q tile of one (b,w,h).
__global__ void k_attn(const _Float16* __restrict__ qkv, float* __restrict__ outp) {
  const int lane = threadIdx.x & 31;
  const int wave = threadIdx.x >> 5;
  const int hf   = lane >> 4;
  const int l15  = lane & 15;
  const int bwh  = blockIdx.y;           // b*NW*H + w*H + h
  const int h    = bwh & (NHEAD-1);
  const int bw   = bwh >> 3;             // b*NW + w
  const int rowbase = bw * NWIN;         // first global row of this window
  const int qTile = blockIdx.x * 4 + wave;   // 0..51
  const float scale = 0.35355339059327373f;  // 1/sqrt(8)

  __shared__ _Float16 lds_p[4][16*32];
  _Float16* pbuf = lds_p[wave];

  // Q fragment (A layout, K padded 8->32 with zeros). dh=8 halves are one
  // aligned 16-byte load; only lanes 0-15 carry real K (0..7).
  v16h qf = {};
  {
    const int nloc = qTile*16 + l15;
    if (nloc < NWIN && hf == 0) {
      const v8h qv = *(const v8h*)(qkv + (rowbase + nloc)*192 + h*DHD);
#pragma unroll
      for (int e = 0; e < 8; ++e) qf[e] = qv[e];
    }
  }

  float mx[8], ls[8];
#pragma unroll
  for (int r = 0; r < 8; ++r) { mx[r] = -1e30f; ls[r] = 0.f; }
  v8f oacc = {};

  for (int j = 0; j < 26; ++j) {          // 26 * 32 = 832 >= 828 columns
    const int cbase = j * 32;
    v8f s[2];
#pragma unroll
    for (int sub = 0; sub < 2; ++sub) {
      const int ncol = cbase + sub*16 + l15;
      const bool okc = ncol < NWIN;
      v16h kf = {};
      if (okc && hf == 0) {
        const v8h kv = *(const v8h*)(qkv + (rowbase + ncol)*192 + CCH + h*DHD);
#pragma unroll
        for (int e = 0; e < 8; ++e) kf[e] = kv[e];
      }
      v8f z = {};
      s[sub] = wmma_f32_f16(qf, kf, z);
      const float maskadd = okc ? 0.f : -1e30f;
#pragma unroll
      for (int r = 0; r < 8; ++r) s[sub][r] = s[sub][r]*scale + maskadd;
    }
    // online softmax over this 32-wide panel (rows = r + 8*hf, cols across lanes)
#pragma unroll
    for (int r = 0; r < 8; ++r) {
      float sm = fmaxf(s[0][r], s[1][r]);
      for (int off = 1; off < 16; off <<= 1) sm = fmaxf(sm, __shfl_xor(sm, off, 32));
      const float newm  = fmaxf(mx[r], sm);
      const float alpha = __expf(mx[r] - newm);
      const float p0 = __expf(s[0][r] - newm);
      const float p1 = __expf(s[1][r] - newm);
      float psum = p0 + p1;
      for (int off = 1; off < 16; off <<= 1) psum += __shfl_xor(psum, off, 32);
      ls[r] = ls[r]*alpha + psum;
      mx[r] = newm;
      oacc[r] *= alpha;
      pbuf[(r + 8*hf)*32 + l15]      = (_Float16)p0;
      pbuf[(r + 8*hf)*32 + 16 + l15] = (_Float16)p1;
    }
    __syncthreads();
    // P tile: C-layout -> A-layout via LDS (paired ds loads)
    v16h pf, vf;
#pragma unroll
    for (int vi = 0; vi < 8; ++vi) {
      const int kk = akk(vi, hf);
      const v2h pp = *(const v2h*)&pbuf[l15*32 + kk];
      pf[2*vi]   = pp[0];
      pf[2*vi+1] = pp[1];
      // V fragment as B (K = window rows of this panel, N = dh padded to 16)
      const int kb = 16*hf + 2*vi;
      const int r0 = cbase + kb, r1 = r0 + 1;
      vf[2*vi]   = (l15 < DHD && r0 < NWIN)
                   ? qkv[(rowbase + r0)*192 + 2*CCH + h*DHD + l15] : (_Float16)0.f;
      vf[2*vi+1] = (l15 < DHD && r1 < NWIN)
                   ? qkv[(rowbase + r1)*192 + 2*CCH + h*DHD + l15] : (_Float16)0.f;
    }
    oacc = wmma_f32_f16(pf, vf, oacc);
    __syncthreads();
  }

  if (l15 < DHD) {
#pragma unroll
    for (int r = 0; r < 8; ++r) {
      const int nloc = qTile*16 + r + 8*hf;
      if (nloc < NWIN)
        outp[(rowbase + nloc)*CCH + h*DHD + l15] = oacc[r] / ls[r];
    }
  }
}

// ---------------- fused FFN: out = relu(X@W1+b1)@W2 + b2 + X -----------------
// one 16-row tile per block; hidden [16,256] kept in LDS as f16.
__global__ void k_ffn(const float* __restrict__ X,
                      const float* __restrict__ W1, const float* __restrict__ b1,
                      const float* __restrict__ W2, const float* __restrict__ b2,
                      float* __restrict__ out) {
  __shared__ _Float16 hbuf[16*256];
  const int lane = threadIdx.x & 31;
  const int wave = threadIdx.x >> 5;
  const int hf   = lane >> 4;
  const int l15  = lane & 15;
  const int m0   = blockIdx.x * 16;
  const int mrow = m0 + l15;

  // A fragments of X (K=64 -> 2 steps), shared across n-tiles
  v16h a[2];
#pragma unroll
  for (int ks = 0; ks < 2; ++ks)
#pragma unroll
    for (int vi = 0; vi < 8; ++vi) {
      const int kk = ks*32 + akk(vi, hf);
      const float2 av = *(const float2*)&X[mrow*CCH + kk];
      a[ks][2*vi]   = (_Float16)av.x;
      a[ks][2*vi+1] = (_Float16)av.y;
    }
  // FFN1: 16 n-tiles over N=256, each wave does 2
#pragma unroll
  for (int tix = 0; tix < 2; ++tix) {
    const int ncol = (wave + tix*8)*16 + l15;
    v8f acc = {};
#pragma unroll
    for (int ks = 0; ks < 2; ++ks) {
      v16h b;
#pragma unroll
      for (int vi = 0; vi < 8; ++vi) {
        const int kb = ks*32 + 16*hf + 2*vi;
        b[2*vi]   = (_Float16)W1[kb*256 + ncol];
        b[2*vi+1] = (_Float16)W1[(kb+1)*256 + ncol];
      }
      acc = wmma_f32_f16(a[ks], b, acc);
    }
    const float bv = b1[ncol];
#pragma unroll
    for (int r = 0; r < 8; ++r)
      hbuf[(r + 8*hf)*256 + ncol] = (_Float16)fmaxf(acc[r] + bv, 0.f);
  }
  __syncthreads();
  // FFN2: N=64 -> 4 n-tiles, waves 0..3; K=256 from LDS (already f16)
  if (wave < 4) {
    const int ncol = wave*16 + l15;
    v8f acc = {};
#pragma unroll
    for (int ks = 0; ks < 8; ++ks) {
      v16h af, b;
#pragma unroll
      for (int vi = 0; vi < 8; ++vi) {
        const int kk = ks*32 + akk(vi, hf);
        const v2h hp = *(const v2h*)&hbuf[l15*256 + kk];
        af[2*vi]   = hp[0];
        af[2*vi+1] = hp[1];
        const int kb = ks*32 + 16*hf + 2*vi;
        b[2*vi]   = (_Float16)W2[kb*CCH + ncol];
        b[2*vi+1] = (_Float16)W2[(kb+1)*CCH + ncol];
      }
      acc = wmma_f32_f16(af, b, acc);
    }
    const float bv = b2[ncol];
#pragma unroll
    for (int r = 0; r < 8; ++r) {
      const int mr = m0 + r + 8*hf;
      out[mr*CCH + ncol] = acc[r] + bv + X[mr*CCH + ncol];
    }
  }
}

// ---------------- per-batch LayerNorm over (C,T,V) + optional roll(T) --------
// xin/out layout: [B,T,V,C]; g,b layout: [C,T,V]; shift: roll(-shift) on T.
__global__ void k_ln(const float* __restrict__ xin, const float* __restrict__ g,
                     const float* __restrict__ bb, float* __restrict__ out, int shift) {
  const int b   = blockIdx.x;
  const int tid = threadIdx.x;
  const int NEL = TT*VV*CCH;     // 211968 (divisible by 1024)
  const float* xb = xin + b*NEL;
  __shared__ float red0[256], red1[256];
  __shared__ float s_mu, s_rstd;
  float s = 0.f, s2 = 0.f;
  for (int i = tid*4; i < NEL; i += 1024) {
    const float4 v4 = *(const float4*)&xb[i];
    s  += v4.x + v4.y + v4.z + v4.w;
    s2 += v4.x*v4.x + v4.y*v4.y + v4.z*v4.z + v4.w*v4.w;
  }
  red0[tid] = s; red1[tid] = s2;
  __syncthreads();
  for (int o = 128; o > 0; o >>= 1) {
    if (tid < o) { red0[tid] += red0[tid+o]; red1[tid] += red1[tid+o]; }
    __syncthreads();
  }
  if (tid == 0) {
    const float mu  = red0[0] / (float)NEL;
    const float var = red1[0] / (float)NEL - mu*mu;
    s_mu = mu; s_rstd = rsqrtf(var + 1e-5f);
  }
  __syncthreads();
  const float mu = s_mu, rs = s_rstd;
  for (int i = tid*4; i < NEL; i += 1024) {
    const int c  = i & (CCH-1);        // c, c+1, c+2, c+3 share (t,v)
    const int tv = i >> 6;
    const int v  = tv % VV;
    const int t  = tv / VV;
    const float4 v4 = *(const float4*)&xb[i];
    float4 y;
    const int gi0 = ((c+0)*TT + t)*VV + v;
    const int gi1 = ((c+1)*TT + t)*VV + v;
    const int gi2 = ((c+2)*TT + t)*VV + v;
    const int gi3 = ((c+3)*TT + t)*VV + v;
    y.x = (v4.x - mu)*rs * g[gi0] + bb[gi0];
    y.y = (v4.y - mu)*rs * g[gi1] + bb[gi1];
    y.z = (v4.z - mu)*rs * g[gi2] + bb[gi2];
    y.w = (v4.w - mu)*rs * g[gi3] + bb[gi3];
    const int tdst = (t + TT - shift) & (TT-1);
    *(float4*)&out[(b*TT*VV + tdst*VV + v)*CCH + c] = y;
  }
}

// ---------------- launcher ---------------------------------------------------
extern "C" void kernel_launch(void* const* d_in, const int* in_sizes, int n_in,
                              void* d_out, int out_size, void* d_ws, size_t ws_size,
                              hipStream_t stream) {
  const float* x = (const float*)d_in[0];
  struct Blk {
    const float *qkv_w, *qkv_b, *proj_w, *proj_b, *w1, *b1, *w2, *b2,
                *g1, *be1, *g2, *be2;
  } blk[2];
  for (int s = 0; s < 2; ++s) {
    const int o = 1 + s*12;
    blk[s].qkv_w  = (const float*)d_in[o+0];
    blk[s].qkv_b  = (const float*)d_in[o+1];
    blk[s].proj_w = (const float*)d_in[o+2];
    blk[s].proj_b = (const float*)d_in[o+3];
    blk[s].w1     = (const float*)d_in[o+4];
    blk[s].b1     = (const float*)d_in[o+5];
    blk[s].w2     = (const float*)d_in[o+6];
    blk[s].b2     = (const float*)d_in[o+7];
    blk[s].g1     = (const float*)d_in[o+8];
    blk[s].be1    = (const float*)d_in[o+9];
    blk[s].g2     = (const float*)d_in[o+10];
    blk[s].be2    = (const float*)d_in[o+11];
  }
  // workspace: cur | tmp | att (f32 [M,64]) | qkv (f16 [M,192])  ~30.5 MB
  float*     cur  = (float*)d_ws;
  float*     tmp  = cur + TOTE;
  float*     att  = tmp + TOTE;
  _Float16*  qkvh = (_Float16*)(att + TOTE);

  k_to_tvc<<<TOTE/256, 256, 0, stream>>>(x, cur);
  for (int s = 0; s < 2; ++s) {
    // QKV projection -> f16 [M, 192]
    k_gemm<CCH, false, true><<<dim3(MROWS/128, 12), 256, 0, stream>>>(
        cur, blk[s].qkv_w, blk[s].qkv_b, nullptr, nullptr, qkvh, 3*CCH);
    // windowed multi-head attention -> f32 [M, 64]
    k_attn<<<dim3(13, BATCH*NWD*NHEAD), 128, 0, stream>>>(qkvh, att);
    // output projection + residual
    k_gemm<CCH, true, false><<<dim3(MROWS/128, 4), 256, 0, stream>>>(
        att, blk[s].proj_w, blk[s].proj_b, cur, tmp, nullptr, CCH);
    // LN1
    k_ln<<<BATCH, 256, 0, stream>>>(tmp, blk[s].g1, blk[s].be1, cur, 0);
    // fused FFN + residual
    k_ffn<<<MROWS/16, 256, 0, stream>>>(cur, blk[s].w1, blk[s].b1,
                                        blk[s].w2, blk[s].b2, tmp);
    // LN2 + cyclic roll by wsz/2 on T
    k_ln<<<BATCH, 256, 0, stream>>>(tmp, blk[s].g2, blk[s].be2, cur, WSZ_/2);
  }
  k_to_ctv<<<TOTE/256, 256, 0, stream>>>(cur, (float*)d_out);
  (void)in_sizes; (void)n_in; (void)out_size; (void)ws_size;
}
